// LidarEncoderMinkUNet_51762945852116
// MI455X (gfx1250) — matrix operand order repaired and checked by reading, not compiled, not run on hardware
//
#include <hip/hip_runtime.h>
#include <hip/hip_bf16.h>

typedef __attribute__((ext_vector_type(16))) __bf16 v16bf;
typedef __attribute__((ext_vector_type(8)))  float  v8f;

#define N_VOX 100000
#define KOFF  27
#define EK    25000
#define BATCH 4
#define R1    (N_VOX + 1)   // +1 dummy row per feature map: zero gather source / zero-add scatter sink

__device__ __forceinline__ __bf16 bf_relu(float v) {
    return (__bf16)fmaxf(v, 0.0f);
}

// ---------------- zero workspace ----------------
__global__ void k_zero(float* __restrict__ p, long n) {
    long i = (long)blockIdx.x * blockDim.x + threadIdx.x;
    long stride = (long)gridDim.x * blockDim.x;
    for (; i < n; i += stride) p[i] = 0.0f;
}

// ---------------- weight pre-swizzle to bf16 (WMMA B layout) ----------------
// W2: [27,32,128] -> [k][tile(8)][lane(32)][j(16)], K index = (j<8?0:8)+(lane<16?0:8)+j, N = tile*16+(lane&15)
__global__ void k_cvtW2(const float* __restrict__ W2, __bf16* __restrict__ out) {
    int i = blockIdx.x * 256 + threadIdx.x;
    if (i >= KOFF * 8 * 32 * 16) return;
    int j = i & 15;
    int L = (i >> 4) & 31;
    int t = (i >> 9) & 7;
    int k = i >> 12;
    int kk = ((j < 8) ? 0 : 8) + ((L < 16) ? 0 : 8) + j;
    int n  = t * 16 + (L & 15);
    out[i] = (__bf16)W2[(k * 32 + kk) * 128 + n];
}

// W3: [27,128,256] -> [k][tile(16)][kchunk(4)][lane(32)][j(16)]
__global__ void k_cvtW3(const float* __restrict__ W3, __bf16* __restrict__ out) {
    int i = blockIdx.x * 256 + threadIdx.x;
    if (i >= KOFF * 16 * 4 * 32 * 16) return;
    int j = i & 15;
    int L = (i >> 4) & 31;
    int q = (i >> 9) & 3;
    int t = (i >> 11) & 15;
    int k = i >> 15;
    int kk = q * 32 + ((j < 8) ? 0 : 8) + ((L < 16) ? 0 : 8) + j;
    int n  = t * 16 + (L & 15);
    out[i] = (__bf16)W3[(k * 128 + kk) * 256 + n];
}

// ---------------- layer 1: Cin=4 -> Cout=32, scalar f32 (tiny) ----------------
__global__ void k_conv1(const float* __restrict__ x0, const float* __restrict__ W1,
                        const int* __restrict__ in_idx, const int* __restrict__ out_idx,
                        float* __restrict__ x1) {
    long gid = (long)blockIdx.x * blockDim.x + threadIdx.x;
    if (gid >= (long)KOFF * EK * 32) return;
    int edge = (int)(gid >> 5);
    int co   = (int)(gid & 31);
    int k    = edge / EK;
    int src  = in_idx[edge];
    int dst  = out_idx[edge];
    const float* xr = x0 + (long)src * 4;
    const float* wr = W1 + k * 128 + co;
    float acc = xr[0] * wr[0] + xr[1] * wr[32] + xr[2] * wr[64] + xr[3] * wr[96];
    atomicAdd(&x1[(long)dst * 32 + co], acc);
}

// ---------------- layer 2: Cin=32 -> Cout=128, bf16 WMMA ----------------
// one wave per (k, 16-edge tile); relu fused into gather; branch-free tail via dummy row R1-1
__global__ void __launch_bounds__(256) k_conv2(const float* __restrict__ x1,
                                               const __bf16* __restrict__ Wc,
                                               const int* __restrict__ in_idx,
                                               const int* __restrict__ out_idx,
                                               float* __restrict__ x2) {
    const int ET = (EK + 15) / 16;
    int wave = (blockIdx.x * blockDim.x + threadIdx.x) >> 5;
    int lane = threadIdx.x & 31;
    if (wave >= KOFF * ET) return;
    int k = wave / ET, et = wave % ET;
    int ebase = et * 16;
    int n0 = lane & 15;

    // A matrix 16x32 bf16: row M = lane&15; lane<16 -> K {0..7,16..23}; lane>=16 -> K {8..15,24..31}
    int mrow = ebase + n0;
    int src  = (mrow < EK) ? in_idx[k * EK + mrow] : N_VOX;   // dummy row gathers zeros
    const float* row = x1 + (long)src * 32;
    int cb = (lane < 16) ? 0 : 8;
    v16bf a;
#pragma unroll
    for (int j = 0; j < 8; ++j) a[j]     = bf_relu(row[cb + j]);
#pragma unroll
    for (int j = 0; j < 8; ++j) a[8 + j] = bf_relu(row[cb + 16 + j]);

    // hoisted scatter pointers: lane holds C/D rows (lane<16 ? 0..7 : 8..15), column n0
    int mo = ebase + ((lane < 16) ? 0 : 8);
    float* p[8];
#pragma unroll
    for (int r = 0; r < 8; ++r) {
        int mm = mo + r;
        int o  = (mm < EK) ? out_idx[k * EK + mm] : N_VOX;    // dummy row: receives exact 0.0 adds
        p[r] = x2 + (long)o * 128 + n0;
    }

#pragma unroll
    for (int t = 0; t < 8; ++t) {
        v16bf b = *(const v16bf*)(Wc + (((long)(k * 8 + t) * 32 + lane) << 4));
        v8f c = {};
        c = __builtin_amdgcn_wmma_f32_16x16x32_bf16(false, a, false, b, (short)0, c, false, false);
#pragma unroll
        for (int r = 0; r < 8; ++r)
            atomicAdd(p[r] + t * 16, c[r]);                   // immediate byte offset t*64
    }
}

// ---------------- layer 3: Cin=128 -> Cout=256, bf16 WMMA, 4 K-chunks, fully unrolled ----------------
__global__ void __launch_bounds__(256) k_conv3(const float* __restrict__ x2,
                                               const __bf16* __restrict__ Wc,
                                               const int* __restrict__ in_idx,
                                               const int* __restrict__ out_idx,
                                               float* __restrict__ x3) {
    const int ET = (EK + 15) / 16;
    int wave = (blockIdx.x * blockDim.x + threadIdx.x) >> 5;
    int lane = threadIdx.x & 31;
    if (wave >= KOFF * ET) return;
    int k = wave / ET, et = wave % ET;
    int ebase = et * 16;
    int n0 = lane & 15;

    int mrow = ebase + n0;
    int src  = (mrow < EK) ? in_idx[k * EK + mrow] : N_VOX;
    const float* row = x2 + (long)src * 128;
    int cb = (lane < 16) ? 0 : 8;
    v16bf a[4];  // A preloaded once, reused across all 16 N-tiles
#pragma unroll
    for (int q = 0; q < 4; ++q) {
#pragma unroll
        for (int j = 0; j < 8; ++j) a[q][j]     = bf_relu(row[q * 32 + cb + j]);
#pragma unroll
        for (int j = 0; j < 8; ++j) a[q][8 + j] = bf_relu(row[q * 32 + cb + 16 + j]);
    }

    int mo = ebase + ((lane < 16) ? 0 : 8);
    float* p[8];
#pragma unroll
    for (int r = 0; r < 8; ++r) {
        int mm = mo + r;
        int o  = (mm < EK) ? out_idx[k * EK + mm] : N_VOX;
        p[r] = x3 + (long)o * 256 + n0;
    }

    const __bf16* wb = Wc + (((long)k * 16) * 4 * 32 + lane) * 16;
#pragma unroll
    for (int t = 0; t < 16; ++t) {
        v8f c = {};
#pragma unroll
        for (int q = 0; q < 4; ++q) {
            v16bf b = *(const v16bf*)(wb + ((long)(t * 4 + q) << 11));  // (t*4+q)*2048 elements, lane folded in base
            c = __builtin_amdgcn_wmma_f32_16x16x32_bf16(false, a[q], false, b, (short)0, c, false, false);
        }
#pragma unroll
        for (int r = 0; r < 8; ++r)
            atomicAdd(p[r] + t * 16, c[r]);                   // immediate byte offset t*64
    }
}

// ---------------- pooled segment-sum of relu(x3) (linearity: project after pooling) ----------------
__global__ void __launch_bounds__(256) k_pool(const float* __restrict__ x3,
                                              const int* __restrict__ batch_idx,
                                              float* __restrict__ S, float* __restrict__ cnt) {
    __shared__ float sS[BATCH * 256];
    __shared__ float scnt[BATCH];
    int tid = threadIdx.x;
    for (int i = tid; i < BATCH * 256; i += 256) sS[i] = 0.0f;
    if (tid < BATCH) scnt[tid] = 0.0f;
    __syncthreads();
    int row0 = blockIdx.x * 128;
    for (int r = 0; r < 128; ++r) {
        int row = row0 + r;
        if (row >= N_VOX) break;
        int b = batch_idx[row];                       // uniform across block
        float v = fmaxf(x3[(long)row * 256 + tid], 0.0f);
        sS[b * 256 + tid] += v;                       // thread owns column tid: race-free
        if (tid == 0) scnt[b] += 1.0f;
    }
    __syncthreads();
    for (int i = tid; i < BATCH * 256; i += 256) atomicAdd(&S[i], sS[i]);
    if (tid < BATCH) atomicAdd(&cnt[tid], scnt[tid]);
}

// ---------------- final tiny GEMM: out[b] = S[b]@proj_w / count + proj_b ----------------
__global__ void k_final(const float* __restrict__ S, const float* __restrict__ cnt,
                        const float* __restrict__ proj_w, const float* __restrict__ proj_b,
                        float* __restrict__ out) {
    int n = threadIdx.x;
    if (n >= 384) return;
#pragma unroll 1
    for (int b = 0; b < BATCH; ++b) {
        float acc = 0.0f;
        for (int c = 0; c < 256; ++c) acc += S[b * 256 + c] * proj_w[c * 384 + n];
        out[b * 384 + n] = acc / fmaxf(cnt[b], 1.0f) + proj_b[n];
    }
}

extern "C" void kernel_launch(void* const* d_in, const int* in_sizes, int n_in,
                              void* d_out, int out_size, void* d_ws, size_t ws_size,
                              hipStream_t stream) {
    const float* x0      = (const float*)d_in[0];
    const float* W1      = (const float*)d_in[1];
    const float* W2      = (const float*)d_in[2];
    const float* W3      = (const float*)d_in[3];
    const float* proj_w  = (const float*)d_in[4];
    const float* proj_b  = (const float*)d_in[5];
    const int*   in_idx  = (const int*)d_in[6];
    const int*   out_idx = (const int*)d_in[7];
    const int*   batch_idx = (const int*)d_in[8];

    float* wsf = (float*)d_ws;
    long off_x1  = 0;                               // (N+1)*32
    long off_x2  = off_x1 + (long)R1 * 32;          // (N+1)*128
    long off_x3  = off_x2 + (long)R1 * 128;         // (N+1)*256
    long off_S   = off_x3 + (long)R1 * 256;
    long off_cnt = off_S + BATCH * 256;
    long zeroN   = off_cnt + 8;                     // pad -> 32B-aligned bf16 region
    float*  x1  = wsf + off_x1;
    float*  x2  = wsf + off_x2;
    float*  x3  = wsf + off_x3;
    float*  S   = wsf + off_S;
    float*  cnt = wsf + off_cnt;
    __bf16* W2c = (__bf16*)(wsf + zeroN);
    __bf16* W3c = W2c + (long)KOFF * 8 * 32 * 16;

    k_zero<<<8192, 256, 0, stream>>>(wsf, zeroN);

    k_cvtW2<<<(KOFF * 8 * 32 * 16 + 255) / 256, 256, 0, stream>>>(W2, W2c);
    k_cvtW3<<<(KOFF * 16 * 4 * 32 * 16 + 255) / 256, 256, 0, stream>>>(W3, W3c);

    long n1 = (long)KOFF * EK * 32;
    k_conv1<<<(int)((n1 + 255) / 256), 256, 0, stream>>>(x0, W1, in_idx, out_idx, x1);

    int ET = (EK + 15) / 16;
    int waves = KOFF * ET;
    int blocks23 = (waves + 7) / 8;                 // 8 waves per 256-thread block
    k_conv2<<<blocks23, 256, 0, stream>>>(x1, W2c, in_idx, out_idx, x2);
    k_conv3<<<blocks23, 256, 0, stream>>>(x2, W3c, in_idx, out_idx, x3);

    k_pool<<<(N_VOX + 127) / 128, 256, 0, stream>>>(x3, batch_idx, S, cnt);
    k_final<<<1, 384, 0, stream>>>(S, cnt, proj_w, proj_b, (float*)d_out);
}
